// LinearAttention_26027501814592
// MI455X (gfx1250) — matrix operand order repaired and checked by reading, not compiled, not run on hardware
//
#include <hip/hip_runtime.h>
#include <stdint.h>

#define N_B   8
#define S_LEN 8192
#define E_DIM 512
#define H_N   8
#define D_H   64
#define M_TOT (N_B * S_LEN)
#define KV_SPLIT 8

typedef __attribute__((ext_vector_type(16))) __bf16 v16bf;
typedef __attribute__((ext_vector_type(8)))  __bf16 v8bf;
typedef __attribute__((ext_vector_type(8)))  float  v8f;
typedef __attribute__((ext_vector_type(8)))  unsigned short v8us;

// ---------- bf16 helpers (bit-level) ----------
__device__ inline unsigned short f2bfu(float f) {
  unsigned u = __float_as_uint(f);
  u += 0x7FFFu + ((u >> 16) & 1u);          // round-to-nearest-even
  return (unsigned short)(u >> 16);
}
__device__ inline float bfu2f(unsigned short s) {
  union { unsigned u; float f; } cv; cv.u = ((unsigned)s) << 16; return cv.f;
}
__device__ inline float bf2f(__bf16 b) {
  union { __bf16 b; unsigned short u; } cv; cv.b = b; return bfu2f(cv.u);
}
__device__ inline v16bf cat8(v8bf l, v8bf h) {
  return __builtin_shufflevector(l, h, 0,1,2,3,4,5,6,7,8,9,10,11,12,13,14,15);
}

// A fragment (16x32 bf16): lane holds row (lane&15); K octets [sel*8,+8) and [16+sel*8,+8)
__device__ inline v16bf load_a_bf16(const unsigned short* base, size_t ld, int row, int k0, int sel) {
  const unsigned short* p = base + (size_t)row * ld + k0 + sel * 8;
  return cat8(*(const v8bf*)p, *(const v8bf*)(p + 16));
}
// B fragment (32x16 bf16): lane holds column (lane&15); K octets [sel*8,+8) and [16+sel*8,+8); memory [col][k]
__device__ inline v16bf load_b_bf16(const unsigned short* base, size_t ld, int col, int k0, int sel) {
  const unsigned short* p = base + (size_t)col * ld + k0 + sel * 8;
  return cat8(*(const v8bf*)p, *(const v8bf*)(p + 16));
}

#define WMMA_BF16(a, b, c) \
  __builtin_amdgcn_wmma_f32_16x16x32_bf16(false, (a), false, (b), (short)0, (c), false, false)

// =====================================================================
// GEMM: Y[M,512] = act(X[M,512] @ W.T + bias); W pre-converted to bf16 [j][c].
// A_BF16: 0 = A is f32, staged+converted through LDS once per block
//         1 = A is bf16, loaded directly
// STORE : 0 = bf16 natural [M,512]; 1 = bf16 head-transposed [n,h,feat%64,s]; 2 = f32 natural
// FMAP  : elu(x)+1  ==  x>0 ? x+1 : exp(x)
// Block = 8 waves; all waves share row tile (blockIdx*64), each owns a 64-wide col tile.
// =====================================================================
template <int A_BF16, int STORE_MODE, int FMAP>
__global__ __launch_bounds__(256) void gemm_k(const void* __restrict__ Ain,
                                              const unsigned short* __restrict__ Wb,
                                              const float* __restrict__ bias,
                                              void* __restrict__ Out) {
  // LDS A slab, fragment-native layout: [buf][k-octet 0..3][row 0..63] x 8 bf16 (16B units)
  __shared__ unsigned short As[2][4 * 64 * 8];
  const int tid  = threadIdx.x;
  const int lane = tid & 31;
  const int wv   = tid >> 5;
  const int row0 = blockIdx.x * 64;
  const int col0 = wv * 64;                 // 8 waves cover N = 512
  const int lo = lane & 15, sel = lane >> 4;

  const v8f vz = {0.f, 0.f, 0.f, 0.f, 0.f, 0.f, 0.f, 0.f};
  v8f c[4][4];
#pragma unroll
  for (int i = 0; i < 4; ++i)
#pragma unroll
    for (int j = 0; j < 4; ++j) c[i][j] = vz;

  const float* Af = (const float*)Ain;
  const unsigned short* Ab = (const unsigned short*)Ain;
  const int arow = tid & 63, aoct = tid >> 6;   // staging role: 4 octets x 64 rows

#pragma unroll 1
  for (int k0 = 0; k0 < E_DIM; k0 += 32) {
    const int buf = (k0 >> 5) & 1;
    if (!A_BF16) {
      // cooperative stage: 8 f32 -> 8 bf16 per thread, one 16B LDS store
      const v8f x = *(const v8f*)(Af + (size_t)(row0 + arow) * E_DIM + k0 + aoct * 8);
      v8us pk;
#pragma unroll
      for (int i = 0; i < 8; ++i) pk[i] = f2bfu(x[i]);
      *(v8us*)(&As[buf][(aoct * 64 + arow) * 8]) = pk;
      __syncthreads();
    }
    v16bf a[4], b[4];
#pragma unroll
    for (int ti = 0; ti < 4; ++ti) {
      if (A_BF16) {
        a[ti] = load_a_bf16(Ab, E_DIM, row0 + ti * 16 + lo, k0, sel);
      } else {
        const v8bf l8 = *(const v8bf*)(&As[buf][(sel * 64 + ti * 16 + lo) * 8]);
        const v8bf h8 = *(const v8bf*)(&As[buf][((2 + sel) * 64 + ti * 16 + lo) * 8]);
        a[ti] = cat8(l8, h8);
      }
    }
#pragma unroll
    for (int tj = 0; tj < 4; ++tj)
      b[tj] = load_b_bf16(Wb, E_DIM, col0 + tj * 16 + lo, k0, sel);
#pragma unroll
    for (int ti = 0; ti < 4; ++ti)
#pragma unroll
      for (int tj = 0; tj < 4; ++tj)
        c[ti][tj] = WMMA_BF16(a[ti], b[tj], c[ti][tj]);
  }

  float bc[4];
#pragma unroll
  for (int tj = 0; tj < 4; ++tj) bc[tj] = bias[col0 + tj * 16 + lo];

#pragma unroll
  for (int ti = 0; ti < 4; ++ti) {
    const int rbase = row0 + ti * 16 + sel * 8;  // 8 consecutive output rows per lane
#pragma unroll
    for (int tj = 0; tj < 4; ++tj) {
      const int colg = col0 + tj * 16 + lo;
      float v[8];
#pragma unroll
      for (int r = 0; r < 8; ++r) {
        float x = c[ti][tj][r] + bc[tj];
        if (FMAP) x = (x > 0.0f) ? (x + 1.0f) : __expf(x);
        v[r] = x;
      }
      if (STORE_MODE == 0) {
        unsigned short* O = (unsigned short*)Out;
#pragma unroll
        for (int r = 0; r < 8; ++r) O[(size_t)(rbase + r) * E_DIM + colg] = f2bfu(v[r]);
      } else if (STORE_MODE == 1) {
        unsigned short* O = (unsigned short*)Out;
        const int n = rbase / S_LEN, s = rbase % S_LEN;   // 8 rows share n (S%64==0)
        const int h = colg / D_H, dd = colg % D_H;
        v8us pk;
#pragma unroll
        for (int r = 0; r < 8; ++r) pk[r] = f2bfu(v[r]);
        *(v8us*)(O + ((size_t)(n * H_N + h) * D_H + dd) * S_LEN + s) = pk;  // 16B store
      } else {
        float* O = (float*)Out;
#pragma unroll
        for (int r = 0; r < 8; ++r) O[(size_t)(rbase + r) * E_DIM + colg] = v[r];
      }
    }
  }
}

// =====================================================================
// KV[nh][m][d] += sum_s v[s,m]*K[s,d] via D[d,m] = KT x VT^T.
// KT,VT are head-transposed bf16 [nh][64][S]: pure b128 loads + WMMA inner loop.
// =====================================================================
__global__ __launch_bounds__(256) void kv_k(const unsigned short* __restrict__ KT,
                                            const unsigned short* __restrict__ VT,
                                            float* __restrict__ KV) {
  const int lane = threadIdx.x & 31, w = threadIdx.x >> 5;
  const int nh = blockIdx.x / KV_SPLIT;
  const int chunk = blockIdx.x % KV_SPLIT;
  const int lo = lane & 15, sel = lane >> 4;
  const int span = S_LEN / KV_SPLIT / 8;  // 128
  const int sBase = chunk * (S_LEN / KV_SPLIT) + w * span;
  const unsigned short* KTh = KT + (size_t)nh * D_H * S_LEN;
  const unsigned short* VTh = VT + (size_t)nh * D_H * S_LEN;

  const v8f vz = {0.f, 0.f, 0.f, 0.f, 0.f, 0.f, 0.f, 0.f};
  v8f c[4][4];
#pragma unroll
  for (int i = 0; i < 4; ++i)
#pragma unroll
    for (int j = 0; j < 4; ++j) c[i][j] = vz;

#pragma unroll 1
  for (int k0 = sBase; k0 < sBase + span; k0 += 32) {
    v16bf a[4], b[4];
#pragma unroll
    for (int ti = 0; ti < 4; ++ti)
      a[ti] = load_a_bf16(KTh, S_LEN, ti * 16 + lo, k0, sel);
#pragma unroll
    for (int tj = 0; tj < 4; ++tj)
      b[tj] = load_b_bf16(VTh, S_LEN, tj * 16 + lo, k0, sel);
#pragma unroll
    for (int ti = 0; ti < 4; ++ti)
#pragma unroll
      for (int tj = 0; tj < 4; ++tj)
        c[ti][tj] = WMMA_BF16(a[ti], b[tj], c[ti][tj]);
  }

#pragma unroll
  for (int ti = 0; ti < 4; ++ti)
#pragma unroll
    for (int tj = 0; tj < 4; ++tj) {
      const int m = tj * 16 + lo, dbase = ti * 16 + sel * 8;
      float* dst = KV + ((size_t)nh * D_H + m) * D_H + dbase;  // [m][d]: contiguous d
#pragma unroll
      for (int r = 0; r < 8; ++r) atomicAdd(dst + r, c[ti][tj][r]);
    }
}

// Ksum[nh*64+d] = sum_s KT[row][s] : one wave per row, shuffle reduction.
__global__ __launch_bounds__(256) void ksum_k(const unsigned short* __restrict__ KT,
                                              float* __restrict__ Ksum) {
  const int lane = threadIdx.x & 31;
  const int gw = blockIdx.x * 8 + (threadIdx.x >> 5);
  const unsigned short* p = KT + (size_t)gw * S_LEN;
  float acc = 0.0f;
#pragma unroll 4
  for (int i = lane * 8; i < S_LEN; i += 32 * 8) {
    const v8bf x = *(const v8bf*)(p + i);
#pragma unroll
    for (int j = 0; j < 8; ++j) acc += bf2f(x[j]);
  }
#pragma unroll
  for (int off = 16; off > 0; off >>= 1) acc += __shfl_down(acc, off, 32);
  if (lane == 0) Ksum[gw] = acc;
}

__global__ void cvt_k(const float* __restrict__ in, unsigned short* __restrict__ out, int n) {
  const int i = blockIdx.x * blockDim.x + threadIdx.x;
  if (i < n) out[i] = f2bfu(in[i]);
}

__global__ void z_k(const unsigned short* __restrict__ Qf,
                    const float* __restrict__ Ksum, float* __restrict__ Z) {
  const int idx = blockIdx.x * blockDim.x + threadIdx.x;
  if (idx >= N_B * S_LEN * H_N) return;
  const int h = idx % H_N;
  const int l = (idx / H_N) % S_LEN;
  const int n = idx / (H_N * S_LEN);
  const unsigned short* q = Qf + ((size_t)n * S_LEN + l) * E_DIM + h * D_H;
  const float* ks = Ksum + (size_t)(n * H_N + h) * D_H;
  float acc = 0.0f;
#pragma unroll 8
  for (int d = 0; d < D_H; ++d) acc += bfu2f(q[d]) * ks[d];
  Z[idx] = 1.0f / (acc + 1e-6f);
}

// =====================================================================
// V[l,m] = Z[l] * (Q[l,:] @ KVbf[m,:]^T)  per (n,h); K-dim = 64 (2 WMMA steps).
// =====================================================================
__global__ __launch_bounds__(256) void attn_k(const unsigned short* __restrict__ Qf,
                                              const unsigned short* __restrict__ KVb,
                                              const float* __restrict__ Z,
                                              unsigned short* __restrict__ Vout) {
  const int lane = threadIdx.x & 31;
  const int gw = blockIdx.x * 8 + (threadIdx.x >> 5);
  const int tilesPerNH = S_LEN / 64;  // 128
  const int nh = gw / tilesPerNH;
  const int rt = gw % tilesPerNH;
  const int n = nh / H_N, h = nh % H_N;
  const int l0 = rt * 64;
  const int lo = lane & 15, sel = lane >> 4;
  const unsigned short* Qb = Qf + (size_t)n * S_LEN * E_DIM + h * D_H;
  const unsigned short* Bb = KVb + (size_t)nh * D_H * D_H;

  const v8f vz = {0.f, 0.f, 0.f, 0.f, 0.f, 0.f, 0.f, 0.f};
  v8f c[4][4];
#pragma unroll
  for (int i = 0; i < 4; ++i)
#pragma unroll
    for (int j = 0; j < 4; ++j) c[i][j] = vz;

#pragma unroll
  for (int k0 = 0; k0 < D_H; k0 += 32) {
    v16bf a[4], b[4];
#pragma unroll
    for (int ti = 0; ti < 4; ++ti)
      a[ti] = load_a_bf16(Qb, E_DIM, l0 + ti * 16 + lo, k0, sel);
#pragma unroll
    for (int tj = 0; tj < 4; ++tj)
      b[tj] = load_b_bf16(Bb, D_H, tj * 16 + lo, k0, sel);
#pragma unroll
    for (int ti = 0; ti < 4; ++ti)
#pragma unroll
      for (int tj = 0; tj < 4; ++tj)
        c[ti][tj] = WMMA_BF16(a[ti], b[tj], c[ti][tj]);
  }

#pragma unroll
  for (int ti = 0; ti < 4; ++ti) {
    const int lbase = l0 + ti * 16 + sel * 8;
    float zv[8];
#pragma unroll
    for (int r = 0; r < 8; ++r) zv[r] = Z[((size_t)n * S_LEN + lbase + r) * H_N + h];
#pragma unroll
    for (int tj = 0; tj < 4; ++tj) {
      const int m = tj * 16 + lo;
#pragma unroll
      for (int r = 0; r < 8; ++r)
        Vout[((size_t)n * S_LEN + lbase + r) * E_DIM + h * D_H + m] =
            f2bfu(c[ti][tj][r] * zv[r]);
    }
  }
}

// =====================================================================
extern "C" void kernel_launch(void* const* d_in, const int* in_sizes, int n_in,
                              void* d_out, int out_size, void* d_ws, size_t ws_size,
                              hipStream_t stream) {
  (void)in_sizes; (void)n_in; (void)out_size; (void)ws_size;
  const float* query = (const float*)d_in[0];
  const float* key_  = (const float*)d_in[1];
  const float* value = (const float*)d_in[2];
  const float* Wq = (const float*)d_in[3];
  const float* bq = (const float*)d_in[4];
  const float* Wk = (const float*)d_in[5];
  const float* bk = (const float*)d_in[6];
  const float* Wv = (const float*)d_in[7];
  const float* bv = (const float*)d_in[8];
  const float* Wo = (const float*)d_in[9];
  const float* bo = (const float*)d_in[10];

  char* w = (char*)d_ws;
  const size_t SZB = (size_t)M_TOT * E_DIM * 2;  // 64 MiB per bf16 activation buffer
  unsigned short* Qf   = (unsigned short*)(w);            // [n,s,e] bf16, feature-mapped Q
  unsigned short* KT   = (unsigned short*)(w + SZB);      // [n,h,d,s] bf16, feature-mapped K^T
  unsigned short* VT   = (unsigned short*)(w + 2 * SZB);  // [n,h,m,s] bf16, v^T
  unsigned short* Vout = (unsigned short*)(w + 3 * SZB);  // [n,s,e] bf16, pre-Wo output
  size_t o = 4 * SZB;
  float* KV           = (float*)(w + o);          o += 1048576;   // [nh][m][d] f32
  float* Ksum         = (float*)(w + o);          o += 16384;     // [nh][d] f32
  unsigned short* KVb = (unsigned short*)(w + o); o += 524288;    // [nh][m][d] bf16
  float* Zb           = (float*)(w + o);          o += 2097152;   // [n,s,h] f32
  unsigned short* Wqb = (unsigned short*)(w + o); o += 524288;    // bf16 weights
  unsigned short* Wkb = (unsigned short*)(w + o); o += 524288;
  unsigned short* Wvb = (unsigned short*)(w + o); o += 524288;
  unsigned short* Wob = (unsigned short*)(w + o); o += 524288;

  const dim3 blk(256);
  const int WSZ = E_DIM * E_DIM;  // 262144
  cvt_k<<<WSZ / 256, blk, 0, stream>>>(Wq, Wqb, WSZ);
  cvt_k<<<WSZ / 256, blk, 0, stream>>>(Wk, Wkb, WSZ);
  cvt_k<<<WSZ / 256, blk, 0, stream>>>(Wv, Wvb, WSZ);
  cvt_k<<<WSZ / 256, blk, 0, stream>>>(Wo, Wob, WSZ);
  hipMemsetAsync(KV, 0, 1048576, stream);  // zero KV for atomics

  gemm_k<0, 0, 1><<<M_TOT / 64, blk, 0, stream>>>(query, Wqb, bq, Qf);
  gemm_k<0, 1, 1><<<M_TOT / 64, blk, 0, stream>>>(key_,  Wkb, bk, KT);
  gemm_k<0, 1, 0><<<M_TOT / 64, blk, 0, stream>>>(value, Wvb, bv, VT);
  ksum_k<<<(N_B * H_N * D_H) / 8, blk, 0, stream>>>(KT, Ksum);
  kv_k<<<N_B * H_N * KV_SPLIT, blk, 0, stream>>>(KT, VT, KV);
  cvt_k<<<(N_B * H_N * D_H * D_H) / 256, blk, 0, stream>>>(KV, KVb, N_B * H_N * D_H * D_H);
  z_k<<<(N_B * S_LEN * H_N) / 256, blk, 0, stream>>>(Qf, Ksum, Zb);
  attn_k<<<(N_B * H_N * (S_LEN / 64)) / 8, blk, 0, stream>>>(Qf, KVb, Zb, Vout);
  gemm_k<1, 2, 0><<<M_TOT / 64, blk, 0, stream>>>(Vout, Wob, bo, d_out);
}
